// MultiHeadAttention_32057635897549
// MI455X (gfx1250) — matrix operand order repaired and checked by reading, not compiled, not run on hardware
//
#include <hip/hip_runtime.h>
#include <hip/hip_bf16.h>

typedef __attribute__((ext_vector_type(16))) __bf16 v16bf;
typedef __attribute__((ext_vector_type(8)))  float  v8f;

#define L_SEQ 1024
#define BATCH 4
#define EMB   1024
#define HEADS 16
#define DHEAD 64
#define NHEAD (BATCH*HEADS)   // 64
#define MROWS (L_SEQ*BATCH)   // 4096

union FragB { v16bf v; uint4 q[2]; __bf16 h[16]; };
union FragC { v8f  v; float f[8]; };

// A-fragment 16x32 bf16: element i of lane -> tile[m][half*16 + i]
__device__ __forceinline__ v16bf loadA(const __bf16* t, int ld, int lane) {
  int m = lane & 15, hf = lane >> 4;
  FragB f;
  f.q[0] = *(const uint4*)(t + m * ld + hf * 16);
  f.q[1] = *(const uint4*)(t + m * ld + hf * 16 + 8);
  return f.v;
}

// B-fragment K32xN16 bf16 from tile stored [n][k] (row stride ld):
// element i of lane -> tile[n][half*16 + i]  (16 contiguous bf16 per lane)
__device__ __forceinline__ v16bf loadB(const __bf16* t, int ld, int lane) {
  int n = lane & 15, hf = lane >> 4;
  const __bf16* p = t + n * ld + hf * 16;
  FragB f;
  f.q[0] = *(const uint4*)p;
  f.q[1] = *(const uint4*)(p + 8);
  return f.v;
}

// stage 16 contiguous f32 -> 16 bf16 into LDS as two b128 stores
__device__ __forceinline__ void cvt_store16(const float* __restrict__ src,
                                            __bf16* __restrict__ dst) {
  float4 s0 = ((const float4*)src)[0];
  float4 s1 = ((const float4*)src)[1];
  float4 s2 = ((const float4*)src)[2];
  float4 s3 = ((const float4*)src)[3];
  __bf16 tmp[16];
  tmp[0]=(__bf16)s0.x; tmp[1]=(__bf16)s0.y; tmp[2]=(__bf16)s0.z; tmp[3]=(__bf16)s0.w;
  tmp[4]=(__bf16)s1.x; tmp[5]=(__bf16)s1.y; tmp[6]=(__bf16)s1.z; tmp[7]=(__bf16)s1.w;
  tmp[8]=(__bf16)s2.x; tmp[9]=(__bf16)s2.y; tmp[10]=(__bf16)s2.z; tmp[11]=(__bf16)s2.w;
  tmp[12]=(__bf16)s3.x; tmp[13]=(__bf16)s3.y; tmp[14]=(__bf16)s3.z; tmp[15]=(__bf16)s3.w;
  ((uint4*)dst)[0] = ((const uint4*)tmp)[0];
  ((uint4*)dst)[1] = ((const uint4*)tmp)[1];
}

// ---------------------------------------------------------------------------
// Kernel 1: q/k/v in-projections.  out = (X @ W^T + b) * sc  -> bf16 [B*H,L,d]
// grid = (M/128, N/64, 3), block = 256 (8 waves)
// ---------------------------------------------------------------------------
__global__ __launch_bounds__(256)
void proj_in_kernel(const float* __restrict__ Xq, const float* __restrict__ Xk,
                    const float* __restrict__ Xv,
                    const float* __restrict__ Wq, const float* __restrict__ Wk,
                    const float* __restrict__ Wv,
                    const float* __restrict__ bq, const float* __restrict__ bk,
                    const float* __restrict__ bv,
                    __bf16* __restrict__ qh, __bf16* __restrict__ kh,
                    __bf16* __restrict__ vh)
{
  __shared__ __bf16 Xl[128 * 32];
  __shared__ __bf16 Wl[64 * 32];

  const int z = blockIdx.z;
  const float* X    = (z == 0) ? Xq : (z == 1) ? Xk : Xv;
  const float* W    = (z == 0) ? Wq : (z == 1) ? Wk : Wv;
  const float* bias = (z == 0) ? bq : (z == 1) ? bk : bv;
  __bf16* out = (z == 0) ? qh : (z == 1) ? kh : vh;
  const float sc = (z == 0) ? 0.125f : 1.0f;     // 1/sqrt(64) only for q

  const int m0 = blockIdx.x * 128, n0 = blockIdx.y * 64;
  const int tid = threadIdx.x, lane = tid & 31, wave = tid >> 5;
  const int wm = wave & 3, wn = wave >> 2;

  FragC acc[2][2];
#pragma unroll
  for (int i = 0; i < 2; ++i)
#pragma unroll
    for (int j = 0; j < 2; ++j)
#pragma unroll
      for (int r = 0; r < 8; ++r) acc[i][j].f[r] = 0.f;

  for (int kc = 0; kc < EMB; kc += 32) {
    __syncthreads();
    {  // stage X tile 128x32 (f32 -> bf16, v_cvt_pk_bf16_f32 path)
      int row = tid >> 1, hc = tid & 1;
      cvt_store16(X + (size_t)(m0 + row) * EMB + kc + hc * 16,
                  Xl + row * 32 + hc * 16);
    }
    if (tid < 128) {  // stage W tile 64x32
      int row = tid >> 1, hc = tid & 1;
      cvt_store16(W + (size_t)(n0 + row) * EMB + kc + hc * 16,
                  Wl + row * 32 + hc * 16);
    }
    __syncthreads();

    v16bf a0 = loadA(Xl + (wm * 32 + 0)  * 32, 32, lane);
    v16bf a1 = loadA(Xl + (wm * 32 + 16) * 32, 32, lane);
    v16bf b0 = loadB(Wl + (wn * 32 + 0)  * 32, 32, lane);
    v16bf b1 = loadB(Wl + (wn * 32 + 16) * 32, 32, lane);

    acc[0][0].v = __builtin_amdgcn_wmma_f32_16x16x32_bf16(false, a0, false, b0, (short)0, acc[0][0].v, false, false);
    acc[0][1].v = __builtin_amdgcn_wmma_f32_16x16x32_bf16(false, a0, false, b1, (short)0, acc[0][1].v, false, false);
    acc[1][0].v = __builtin_amdgcn_wmma_f32_16x16x32_bf16(false, a1, false, b0, (short)0, acc[1][0].v, false, false);
    acc[1][1].v = __builtin_amdgcn_wmma_f32_16x16x32_bf16(false, a1, false, b1, (short)0, acc[1][1].v, false, false);
  }

  const int hf = lane >> 4, col = lane & 15;
#pragma unroll
  for (int i = 0; i < 2; ++i)
#pragma unroll
    for (int j = 0; j < 2; ++j) {
      int n = n0 + wn * 32 + j * 16 + col;
      float bn = bias[n];
      int hh = n >> 6, dd = n & 63;
#pragma unroll
      for (int r = 0; r < 8; ++r) {
        int m = m0 + wm * 32 + i * 16 + r + 8 * hf;
        int l = m >> 2, b = m & 3;               // m = l*B + b, B=4
        float v = (acc[i][j].f[r] + bn) * sc;
        out[(((size_t)(b * HEADS + hh)) * L_SEQ + l) * DHEAD + dd] = (__bf16)v;
      }
    }
}

// ---------------------------------------------------------------------------
// Kernel 2: fused gated attention, one pass over pos-weights.
// Clipping bounds (content<=10, log_sigmoid<=0) make w<=10 always, so softmax
// uses the FIXED shift 10: p = exp(w-10) = exp(clip(content)-10)*sigmoid(pos).
// -> no running max, no accumulator rescale, lane-private partial row sums
//    reduced across lanes only once at the end.
// grid = (L/128, B*H), block = 256 (8 waves, 16 query rows per wave)
// K tile staged via CDNA5 global_load_async_to_lds_b128 (ASYNCcnt path).
// ---------------------------------------------------------------------------
__global__ __launch_bounds__(256)
void attn_kernel(const __bf16* __restrict__ qh,
                 const __bf16* __restrict__ kh,
                 const __bf16* __restrict__ vh,
                 const float* __restrict__ pw,
                 __bf16* __restrict__ oh)   // bf16 [L,B,E]
{
  __shared__ __bf16 Kl[32 * 64];   // [key][d]
  __shared__ __bf16 Vt[64 * 32];   // [d][key]  (transposed at stage)
  __shared__ __bf16 Pl[8 * 16 * 32];

  const int head = blockIdx.y;             // n = b*H + h
  const int tid = threadIdx.x, lane = tid & 31, wave = tid >> 5;
  const int qbase = blockIdx.x * 128 + wave * 16;
  const int hf = lane >> 4, col = lane & 15;

  const __bf16* Q = qh + (size_t)head * L_SEQ * DHEAD;
  const __bf16* K = kh + (size_t)head * L_SEQ * DHEAD;
  const __bf16* V = vh + (size_t)head * L_SEQ * DHEAD;
  const float* P = pw + (size_t)head * L_SEQ * L_SEQ;

  // Q A-fragments (16 rows x d=64, two K32 chunks), direct from global
  v16bf aQ0 = loadA(Q + (size_t)qbase * DHEAD + 0,  DHEAD, lane);
  v16bf aQ1 = loadA(Q + (size_t)qbase * DHEAD + 32, DHEAD, lane);

  FragC acc[4];
  float lrow[8];                          // lane-private partial row sums
#pragma unroll
  for (int j = 0; j < 4; ++j)
#pragma unroll
    for (int r = 0; r < 8; ++r) acc[j].f[r] = 0.f;
#pragma unroll
  for (int r = 0; r < 8; ++r) lrow[r] = 0.f;

  __bf16* Pw = Pl + wave * 16 * 32;
  const int srow = tid >> 3, sch = tid & 7;              // staging assignment
  const unsigned kl_off = (unsigned)(size_t)(Kl + srow * 64 + sch * 8);

  for (int kb = 0; kb < L_SEQ; kb += 32) {
    __syncthreads();
    {  // K tile: async DMA to LDS (16B per thread, 4KB total)
      const __bf16* gk = K + (size_t)(kb + srow) * DHEAD + sch * 8;
      asm volatile("global_load_async_to_lds_b128 %0, %1, off"
                   :: "v"(kl_off), "v"(gk) : "memory");
      // V tile: load + transpose into LDS
      uint4 vvv = *(const uint4*)(V + (size_t)(kb + srow) * DHEAD + sch * 8);
      const __bf16* vs = (const __bf16*)&vvv;
#pragma unroll
      for (int j = 0; j < 8; ++j) Vt[(sch * 8 + j) * 32 + srow] = vs[j];
      asm volatile("s_wait_asynccnt 0x0" ::: "memory");
    }
    __syncthreads();

    if (kb + 32 < L_SEQ)   // hint next pos-weight block into cache
      __builtin_prefetch(P + (size_t)(qbase + col) * L_SEQ + kb + 32, 0, 1);

#pragma unroll
    for (int nb = 0; nb < 2; ++nb) {
      v16bf bk0 = loadB(Kl + (nb * 16) * 64 + 0,  64, lane);
      v16bf bk1 = loadB(Kl + (nb * 16) * 64 + 32, 64, lane);
      FragC s;
#pragma unroll
      for (int r = 0; r < 8; ++r) s.f[r] = 0.f;
      s.v = __builtin_amdgcn_wmma_f32_16x16x32_bf16(false, aQ0, false, bk0, (short)0, s.v, false, false);
      s.v = __builtin_amdgcn_wmma_f32_16x16x32_bf16(false, aQ1, false, bk1, (short)0, s.v, false, false);
#pragma unroll
      for (int r = 0; r < 8; ++r) {
        int m = r + 8 * hf;
        float c = fminf(fmaxf(s.f[r], -10.f), 10.f);
        float x = P[(size_t)(qbase + m) * L_SEQ + kb + nb * 16 + col];
        x = fminf(fmaxf(x, -10.f), 10.f);
        // p = exp(log_sigmoid(x) + c - 10) = exp(c-10) / (1 + exp(-x))
        float p = __fdividef(__expf(c - 10.f), 1.f + __expf(-x));
        lrow[r] += p;
        Pw[m * 32 + nb * 16 + col] = (__bf16)p;
      }
    }

    v16bf aP = loadA(Pw, 32, lane);   // same-wave LDS RAW: dscnt handled by compiler
#pragma unroll
    for (int j = 0; j < 4; ++j) {
      v16bf bvf = loadB(Vt + (j * 16) * 32, 32, lane);
      acc[j].v = __builtin_amdgcn_wmma_f32_16x16x32_bf16(false, aP, false, bvf, (short)0, acc[j].v, false, false);
    }
  }

  // final row-sum reduction across the 16 lanes of each half, then normalize
  const int b = head >> 4, hh = head & 15;
#pragma unroll
  for (int r = 0; r < 8; ++r) {
    float t = lrow[r];
#pragma unroll
    for (int o = 8; o >= 1; o >>= 1) t += __shfl_xor(t, o, 32);
    float inv = __fdividef(1.f, t);
    int l = qbase + r + 8 * hf;
    size_t base = ((size_t)l * BATCH + b) * EMB + hh * DHEAD;
#pragma unroll
    for (int j = 0; j < 4; ++j)
      oh[base + j * 16 + col] = (__bf16)(acc[j].f[r] * inv);
  }
}

// ---------------------------------------------------------------------------
// Kernel 3: out-projection.  out = O @ Wo^T + bo  (O bf16, out fp32)
// grid = (M/128, N/64), block = 256
// ---------------------------------------------------------------------------
__global__ __launch_bounds__(256)
void proj_out_kernel(const __bf16* __restrict__ O,
                     const float* __restrict__ Wo, const float* __restrict__ bo,
                     float* __restrict__ out)
{
  __shared__ __bf16 Xl[128 * 32];
  __shared__ __bf16 Wl[64 * 32];

  const int m0 = blockIdx.x * 128, n0 = blockIdx.y * 64;
  const int tid = threadIdx.x, lane = tid & 31, wave = tid >> 5;
  const int wm = wave & 3, wn = wave >> 2;

  FragC acc[2][2];
#pragma unroll
  for (int i = 0; i < 2; ++i)
#pragma unroll
    for (int j = 0; j < 2; ++j)
#pragma unroll
      for (int r = 0; r < 8; ++r) acc[i][j].f[r] = 0.f;

  for (int kc = 0; kc < EMB; kc += 32) {
    __syncthreads();
    {  // X already bf16: straight 32B copies
      int row = tid >> 1, hc = tid & 1;
      const uint4* s = (const uint4*)(O + (size_t)(m0 + row) * EMB + kc + hc * 16);
      uint4* d = (uint4*)(Xl + row * 32 + hc * 16);
      d[0] = s[0]; d[1] = s[1];
    }
    if (tid < 128) {
      int row = tid >> 1, hc = tid & 1;
      cvt_store16(Wo + (size_t)(n0 + row) * EMB + kc + hc * 16,
                  Wl + row * 32 + hc * 16);
    }
    __syncthreads();

    v16bf a0 = loadA(Xl + (wm * 32 + 0)  * 32, 32, lane);
    v16bf a1 = loadA(Xl + (wm * 32 + 16) * 32, 32, lane);
    v16bf b0 = loadB(Wl + (wn * 32 + 0)  * 32, 32, lane);
    v16bf b1 = loadB(Wl + (wn * 32 + 16) * 32, 32, lane);

    acc[0][0].v = __builtin_amdgcn_wmma_f32_16x16x32_bf16(false, a0, false, b0, (short)0, acc[0][0].v, false, false);
    acc[0][1].v = __builtin_amdgcn_wmma_f32_16x16x32_bf16(false, a0, false, b1, (short)0, acc[0][1].v, false, false);
    acc[1][0].v = __builtin_amdgcn_wmma_f32_16x16x32_bf16(false, a1, false, b0, (short)0, acc[1][0].v, false, false);
    acc[1][1].v = __builtin_amdgcn_wmma_f32_16x16x32_bf16(false, a1, false, b1, (short)0, acc[1][1].v, false, false);
  }

  const int hf = lane >> 4, col = lane & 15;
#pragma unroll
  for (int i = 0; i < 2; ++i)
#pragma unroll
    for (int j = 0; j < 2; ++j) {
      int n = n0 + wn * 32 + j * 16 + col;
      float bn = bo[n];
#pragma unroll
      for (int r = 0; r < 8; ++r) {
        int m = m0 + wm * 32 + i * 16 + r + 8 * hf;
        out[(size_t)m * EMB + n] = acc[i][j].f[r] + bn;
      }
    }
}

// ---------------------------------------------------------------------------
extern "C" void kernel_launch(void* const* d_in, const int* in_sizes, int n_in,
                              void* d_out, int out_size, void* d_ws, size_t ws_size,
                              hipStream_t stream) {
  const float* query = (const float*)d_in[0];
  const float* key   = (const float*)d_in[1];
  const float* value = (const float*)d_in[2];
  const float* pw    = (const float*)d_in[3];
  const float* Wq = (const float*)d_in[4];
  const float* bq = (const float*)d_in[5];
  const float* Wk = (const float*)d_in[6];
  const float* bk = (const float*)d_in[7];
  const float* Wv = (const float*)d_in[8];
  const float* bv = (const float*)d_in[9];
  const float* Wo = (const float*)d_in[10];
  const float* bo = (const float*)d_in[11];

  // workspace layout: qh | kh | vh  in [B*H, L, d] bf16, oh in [L,B,E] bf16
  const size_t headElems = (size_t)NHEAD * L_SEQ * DHEAD;   // 4M elems each
  __bf16* qh = (__bf16*)d_ws;
  __bf16* kh = qh + headElems;
  __bf16* vh = kh + headElems;
  __bf16* oh = vh + headElems;                              // 4M elems

  dim3 gA(MROWS / 128, EMB / 64, 3);
  proj_in_kernel<<<gA, 256, 0, stream>>>(query, key, value, Wq, Wk, Wv,
                                         bq, bk, bv, qh, kh, vh);

  dim3 gB(L_SEQ / 128, NHEAD);
  attn_kernel<<<gB, 256, 0, stream>>>(qh, kh, vh, pw, oh);

  dim3 gC(MROWS / 128, EMB / 64);
  proj_out_kernel<<<gC, 256, 0, stream>>>(oh, Wo, bo, (float*)d_out);
}
